// SurpriseTracker_39986145526015
// MI455X (gfx1250) — matrix operand order repaired and checked by reading, not compiled, not run on hardware
//
#include <hip/hip_runtime.h>
#include <hip/hip_bf16.h>
#include <math.h>

// Problem constants (fixed by setup_inputs)
#define BB 4
#define LL 8192
#define CC 1024
#define HH 256
#define ROWS (BB * LL)          // 32768 token rows
#define KSTEPS (CC / 32)        // 32 WMMA k-steps of K=32
#define NTILES (HH / 16)        // 16 N-tiles of 16
#define TOK_PER_WG 16           // one 16-token tile per block
#define LDS_STRIDE 1032         // 1024 + 8 bf16 pad (row base stays 16B aligned)

typedef __attribute__((ext_vector_type(16))) __bf16 v16bf;
typedef __attribute__((ext_vector_type(8)))  float  v8f;

static __device__ __forceinline__ unsigned short f2bf(float f) {
    unsigned int u = __float_as_uint(f);
    u += 0x7fffu + ((u >> 16) & 1u);   // round-to-nearest-even
    return (unsigned short)(u >> 16);
}
static __device__ __forceinline__ float sigmoidf_(float z) { return 1.0f / (1.0f + __expf(-z)); }
static __device__ __forceinline__ float siluf_(float z)    { return z * sigmoidf_(z); }

// ---------------------------------------------------------------------------
// Kernel 0: pack w1e/w1t (f32 [C][H]) into bf16 WMMA B-tile layout.
// Flat index = ((((gate*16+nt)*32+kstep)*32+lane)*16+j)
// lane holds 16 contiguous bf16 = B[k..k+15][col] for its half of K.
// ---------------------------------------------------------------------------
__global__ void st_prepack(const float* __restrict__ w1e,
                           const float* __restrict__ w1t,
                           unsigned short* __restrict__ wpack) {
    int idx = blockIdx.x * blockDim.x + threadIdx.x;   // 0 .. 2*C*H-1 = 524287
    int j     =  idx        & 15;
    int lane  = (idx >> 4)  & 31;
    int kstep = (idx >> 9)  & 31;
    int nt    = (idx >> 14) & 15;
    int gate  = (idx >> 18) & 1;
    const float* w1 = gate ? w1t : w1e;
    int col = nt * 16 + (lane & 15);
    int k   = kstep * 32 + ((lane >> 4) << 4) + j;
    wpack[idx] = f2bf(w1[k * HH + col]);
}

// ---------------------------------------------------------------------------
// Kernel 1: gates. Block = 256 threads (8 waves), 16 tokens per block.
// x row-block staged once in LDS as bf16; the 32 (gate,nt) combos are split
// 4-per-wave so only 4 accumulator tiles (32 VGPRs) are live -> no spills.
// A tile loaded once per k-step (2x ds_load_b128) and reused by 4 WMMAs.
// B addressing: one per-lane base pointer advanced 1KB/k-step; the 4 combos
// sit at fixed 32KB offsets -> global_load_b128 immediate offsets, no per-load
// 64-bit address math.
// ---------------------------------------------------------------------------
__global__ void st_gates(const float* __restrict__ x,
                         const unsigned short* __restrict__ wpack,
                         const float* __restrict__ b1e, const float* __restrict__ w2e,
                         const float* __restrict__ b2e,
                         const float* __restrict__ b1t, const float* __restrict__ w2t,
                         const float* __restrict__ b2t,
                         float* __restrict__ etaArr, float* __restrict__ thetaArr) {
    __shared__ unsigned short tile[TOK_PER_WG * LDS_STRIDE];  // 33 KB
    __shared__ float red[32][16];                             // per-combo token partials

    const int tid  = threadIdx.x;
    const int lane = tid & 31;
    const int wave = tid >> 5;
    const int rowBase = blockIdx.x * TOK_PER_WG;

    const int laneLo = lane & 15;
    const int laneHi = lane >> 4;

    // ---- stage x (16 rows x 1024 cols) as bf16 into LDS, once ----
    for (int i = tid; i < TOK_PER_WG * (CC / 4); i += 256) {
        int r  = i >> 8;              // 256 float4 groups per row
        int c4 = (i & 255) << 2;
        const float4 v = *(const float4*)(x + (size_t)(rowBase + r) * CC + c4);
        *(ushort4*)&tile[r * LDS_STRIDE + c4] =
            make_ushort4(f2bf(v.x), f2bf(v.y), f2bf(v.z), f2bf(v.w));
    }
    __syncthreads();

    // ---- wave handles combos wave*4 .. wave*4+3 ; combo = gate*16 + nt ----
    const int combo0 = wave * 4;      // waves 0-3 -> gate 0, waves 4-7 -> gate 1
    const int gate   = combo0 >> 4;
    const int nt0    = combo0 & 15;

    // per-lane bases, hoisted out of the k loop
    const unsigned short* bptr =
        wpack + (((size_t)combo0 * 32) * 32 + lane) * 16;     // +512 elem / k-step
    const unsigned short* aptr = &tile[laneLo * LDS_STRIDE + (laneHi << 4)];

    v8f acc[4];
#pragma unroll
    for (int i = 0; i < 4; ++i) acc[i] = (v8f){0,0,0,0,0,0,0,0};

    for (int kstep = 0; kstep < KSTEPS; ++kstep) {
        // A tile (16 tokens x 32 K), bf16 layout, from LDS
        union { v16bf v; uint4 q[2]; } A;
        A.q[0] = *(const uint4*)(aptr);
        A.q[1] = *(const uint4*)(aptr + 8);
        aptr += 32;

        // prefetch next k-step's B stream (global_prefetch_b8)
        __builtin_prefetch(bptr + 512, 0, 1);

        const v16bf* bp = (const v16bf*)bptr;   // combos at 1024 v16bf (32KB) strides
        acc[0] = __builtin_amdgcn_wmma_f32_16x16x32_bf16(
            false, A.v, false, bp[0],    (short)0, acc[0], false, false);
        acc[1] = __builtin_amdgcn_wmma_f32_16x16x32_bf16(
            false, A.v, false, bp[1024], (short)0, acc[1], false, false);
        acc[2] = __builtin_amdgcn_wmma_f32_16x16x32_bf16(
            false, A.v, false, bp[2048], (short)0, acc[2], false, false);
        acc[3] = __builtin_amdgcn_wmma_f32_16x16x32_bf16(
            false, A.v, false, bp[3072], (short)0, acc[3], false, false);
        bptr += 512;
    }

    // ---- layer 2 partials: silu(h + b1) * w2, reduced across 16 N columns ----
    const float* b1 = gate ? b1t : b1e;
    const float* w2 = gate ? w2t : w2e;
#pragma unroll
    for (int i = 0; i < 4; ++i) {
        const int nt = nt0 + i;
        const float b1c = b1[nt * 16 + laneLo];
        const float w2c = w2[nt * 16 + laneLo];
        float part[8];
#pragma unroll
        for (int r = 0; r < 8; ++r)
            part[r] = siluf_(acc[i][r] + b1c) * w2c;
        // sum the 16 N-columns held across each half-wave (fixed order)
#pragma unroll
        for (int m = 1; m < 16; m <<= 1) {
#pragma unroll
            for (int r = 0; r < 8; ++r)
                part[r] += __shfl_xor(part[r], m, 32);
        }
        if (laneLo == 0) {
#pragma unroll
            for (int r = 0; r < 8; ++r)
                red[combo0 + i][laneHi * 8 + r] = part[r];
        }
    }
    __syncthreads();

    // ---- final: sum over nt (fixed order), sigmoid, store eta/theta ----
    if (tid < 32) {
        const int g     = tid >> 4;
        const int token = tid & 15;
        float s = 0.0f;
#pragma unroll
        for (int nt = 0; nt < NTILES; ++nt) s += red[g * 16 + nt][token];
        const float b2 = g ? b2t[0] : b2e[0];
        float* go = g ? thetaArr : etaArr;
        go[rowBase + token] = sigmoidf_(s + b2);
    }
}

// ---------------------------------------------------------------------------
// Kernel 2: per-row L2 norm of model_output. One wave per row.
// ---------------------------------------------------------------------------
__global__ void st_norm(const float* __restrict__ mo, float* __restrict__ m_raw) {
    const int lane = threadIdx.x & 31;
    const int wave = threadIdx.x >> 5;
    const size_t row = (size_t)blockIdx.x * 8 + wave;
    const float4* p = (const float4*)(mo + row * CC);
    float s = 0.0f;
#pragma unroll
    for (int i = 0; i < 8; ++i) {
        float4 v = p[lane + i * 32];
        s += v.x * v.x + v.y * v.y + v.z * v.z + v.w * v.w;
    }
#pragma unroll
    for (int m = 16; m >= 1; m >>= 1) s += __shfl_xor(s, m, 32);
    if (lane == 0) m_raw[row] = sqrtf(s);
}

// Kernel 3: mean of m over L per batch (fixed-order tree -> deterministic).
__global__ void st_mean(const float* __restrict__ m_raw, float* __restrict__ meanArr) {
    __shared__ float sm[1024];
    const int b = blockIdx.x, tid = threadIdx.x;
    float s = 0.0f;
#pragma unroll
    for (int k = 0; k < LL / 1024; ++k) s += m_raw[(size_t)b * LL + k * 1024 + tid];
    sm[tid] = s; __syncthreads();
    for (int off = 512; off >= 1; off >>= 1) {
        if (tid < off) sm[tid] += sm[tid + off];
        __syncthreads();
    }
    if (tid == 0) meanArr[b] = sm[0] / (float)LL;
}

// ---------------------------------------------------------------------------
// Kernel 4: affine scan per batch. s_t = eta_t*s_{t-1} - theta_t*m_t
// => (P_t, U_t) inclusive prefix of affine maps; block = 1024 thr, 8 t/thr.
// ---------------------------------------------------------------------------
__global__ void st_scan(const float* __restrict__ etaArr, const float* __restrict__ thetaArr,
                        const float* __restrict__ m_raw, const float* __restrict__ meanArr,
                        float* __restrict__ P, float* __restrict__ U) {
    __shared__ float sA[1024], sB[1024];
    const int b = blockIdx.x, tid = threadIdx.x;
    const float inv = 1.0f / (meanArr[b] + 1e-8f);
    const size_t base = (size_t)b * LL + (size_t)tid * 8;

    float ea[8], tm[8];
    float A = 1.0f, Bv = 0.0f;
#pragma unroll
    for (int j = 0; j < 8; ++j) {
        float e = etaArr[base + j];
        float t = thetaArr[base + j];
        float m = m_raw[base + j] * inv;
        ea[j] = e; tm[j] = t * m;
        A = e * A; Bv = e * Bv - t * m;
    }
    float iA = A, iB = Bv;                     // inclusive prefix across threads
    for (int off = 1; off < 1024; off <<= 1) {
        sA[tid] = iA; sB[tid] = iB; __syncthreads();
        float pA = 1.0f, pB = 0.0f;
        if (tid >= off) { pA = sA[tid - off]; pB = sB[tid - off]; }
        __syncthreads();
        float cA = iA, cB = iB;
        iA = cA * pA; iB = cA * pB + cB;       // compose prev-then-cur
    }
    sA[tid] = iA; sB[tid] = iB; __syncthreads();
    float eAp = 1.0f, eBp = 0.0f;              // exclusive prefix
    if (tid > 0) { eAp = sA[tid - 1]; eBp = sB[tid - 1]; }
#pragma unroll
    for (int j = 0; j < 8; ++j) {
        float nP = ea[j] * eAp;
        float nU = ea[j] * eBp - tm[j];
        P[base + j] = nP; U[base + j] = nU;
        eAp = nP; eBp = nU;
    }
}

// Kernel 5: out[row, c] = P[row]*past_surprise[c] + U[row]   (float4)
__global__ void st_out(const float* __restrict__ P, const float* __restrict__ U,
                       const float* __restrict__ ps, float* __restrict__ out) {
    const size_t idx = (size_t)blockIdx.x * blockDim.x + threadIdx.x; // ROWS*256
    const size_t row = idx >> 8;
    const int    c4  = (int)(idx & 255) << 2;
    const float4 p = *(const float4*)(ps + c4);
    const float Pv = P[row], Uv = U[row];
    float4 o;
    o.x = Pv * p.x + Uv; o.y = Pv * p.y + Uv;
    o.z = Pv * p.z + Uv; o.w = Pv * p.w + Uv;
    *(float4*)(out + row * CC + c4) = o;
}

// ---------------------------------------------------------------------------
extern "C" void kernel_launch(void* const* d_in, const int* in_sizes, int n_in,
                              void* d_out, int out_size, void* d_ws, size_t ws_size,
                              hipStream_t stream) {
    const float* x    = (const float*)d_in[0];
    const float* mo   = (const float*)d_in[1];
    const float* ps   = (const float*)d_in[2];
    const float* w1e  = (const float*)d_in[3];
    const float* b1e  = (const float*)d_in[4];
    const float* w2e  = (const float*)d_in[5];
    const float* b2e  = (const float*)d_in[6];
    const float* w1t  = (const float*)d_in[7];
    const float* b1t  = (const float*)d_in[8];
    const float* w2t  = (const float*)d_in[9];
    const float* b2t  = (const float*)d_in[10];
    float* out = (float*)d_out;

    char* ws = (char*)d_ws;
    unsigned short* wpack = (unsigned short*)(ws);              // 1 MB (2*C*H bf16)
    float* etaArr  = (float*)(ws + 1048576);                    // 128 KB
    float* thetaArr= (float*)(ws + 1048576 + 131072);
    float* m_raw   = (float*)(ws + 1048576 + 2 * 131072);
    float* meanArr = (float*)(ws + 1048576 + 3 * 131072);       // 16 B (+pad)
    float* Pbuf    = (float*)(ws + 1048576 + 3 * 131072 + 256);
    float* Ubuf    = (float*)(ws + 1048576 + 3 * 131072 + 256 + 131072);

    st_prepack<<<(2 * CC * HH) / 256, 256, 0, stream>>>(w1e, w1t, wpack);
    st_norm<<<ROWS / 8, 256, 0, stream>>>(mo, m_raw);
    st_mean<<<BB, 1024, 0, stream>>>(m_raw, meanArr);
    st_gates<<<ROWS / TOK_PER_WG, 256, 0, stream>>>(x, wpack,
                                                    b1e, w2e, b2e,
                                                    b1t, w2t, b2t,
                                                    etaArr, thetaArr);
    st_scan<<<BB, 1024, 0, stream>>>(etaArr, thetaArr, m_raw, meanArr, Pbuf, Ubuf);
    st_out<<<(ROWS * (CC / 4)) / 256, 256, 0, stream>>>(Pbuf, Ubuf, ps, out);
}
